// SelfAttention_54915451847174
// MI455X (gfx1250) — matrix operand order repaired and checked by reading, not compiled, not run on hardware
//
#include <hip/hip_runtime.h>

typedef __attribute__((ext_vector_type(16))) _Float16     v16h;
typedef __attribute__((ext_vector_type(8)))  _Float16     v8h;
typedef __attribute__((ext_vector_type(8)))  float        v8f;
typedef __attribute__((ext_vector_type(4)))  unsigned int v4u;
typedef __attribute__((ext_vector_type(8)))  int          v8i;
typedef __attribute__((ext_vector_type(4)))  int          v4i;

#define D_MODEL 1024
#define DIMS    128
#define SEQ     4096
#define BATCH   4
#define N3      384                     // q|k|v concatenated output channels
#define SCALE   0.08838834764831845f    // 128^-0.5
#define SPLITK  8
#define TCHUNK  32
#define SROW    136                     // padded LDS row stride in halves (272B)

static __device__ __forceinline__ v16h cat8(v8h lo, v8h hi) {
  return __builtin_shufflevector(lo, hi, 0,1,2,3,4,5,6,7,8,9,10,11,12,13,14,15);
}

// ---------------------------------------------------------------------------
// Kernel 1: build f16 Wcat[384][1024] (rows: Wq|Wk|Wv), f32 bias cat, zero Sacc
// ---------------------------------------------------------------------------
__global__ void prep_kernel(const float* __restrict__ Wq, const float* __restrict__ Wk,
                            const float* __restrict__ Wv, const float* __restrict__ bq,
                            const float* __restrict__ bk, const float* __restrict__ bv,
                            _Float16* __restrict__ Wcat, float* __restrict__ bcat,
                            float* __restrict__ Sacc) {
  int gid = blockIdx.x * blockDim.x + threadIdx.x;
  if (gid < N3 * D_MODEL) {
    int n = gid / D_MODEL, m = gid % D_MODEL;
    float w = (n < DIMS)   ? Wq[n * D_MODEL + m]
            : (n < 2*DIMS) ? Wk[(n - DIMS) * D_MODEL + m]
                           : Wv[(n - 2*DIMS) * D_MODEL + m];
    Wcat[gid] = (_Float16)w;
  }
  if (gid < N3) {
    bcat[gid] = (gid < DIMS) ? bq[gid] : (gid < 2*DIMS) ? bk[gid - DIMS] : bv[gid - 2*DIMS];
  }
  if (gid < BATCH * DIMS * DIMS) Sacc[gid] = 0.0f;
}

// ---------------------------------------------------------------------------
// Kernel 2: QKV projection GEMM  [16384 x 1024] * [1024 x 384] -> f16 QKV
// 256 threads = 8 waves; block computes 32 rows x all 384 cols.
// Wave w owns N-tiles {3w, 3w+1, 3w+2}; x tile staged via LDS (f32->f16).
// ---------------------------------------------------------------------------
__global__ void __launch_bounds__(256) qkv_kernel(const float* __restrict__ x,
                                                  const _Float16* __restrict__ Wcat,
                                                  const float* __restrict__ bcat,
                                                  _Float16* __restrict__ QKV) {
  __shared__ _Float16 ldsA[32][40];           // 32 rows x 32 K, stride 40 (16B-aligned rows)
  const int tid  = threadIdx.x;
  const int lane = tid & 31;
  const int wave = tid >> 5;
  const int r0   = blockIdx.x * 32;           // row into flattened [B*S]
  const int rowA = lane & 15;
  const int koff = (lane < 16) ? 0 : 8;       // A-frag K split per ISA layout
  const int nb0  = wave * 48;
  const int srow = tid >> 3;                  // staging row 0..31
  const int sc4  = (tid & 7) * 4;             // staging col 0..28

  v8f acc[2][3] = {};

  for (int kt = 0; kt < D_MODEL; kt += 32) {
    {
      const float4 f = *(const float4*)(x + (size_t)(r0 + srow) * D_MODEL + kt + sc4);
      // prefetch next K-tile of x (global_prefetch_b8) to hide HBM latency
      if (kt + 32 < D_MODEL)
        __builtin_prefetch(x + (size_t)(r0 + srow) * D_MODEL + kt + 32 + sc4, 0, 0);
      ldsA[srow][sc4 + 0] = (_Float16)f.x;
      ldsA[srow][sc4 + 1] = (_Float16)f.y;
      ldsA[srow][sc4 + 2] = (_Float16)f.z;
      ldsA[srow][sc4 + 3] = (_Float16)f.w;
    }
    __syncthreads();

    v16h afrag[2];
#pragma unroll
    for (int ms = 0; ms < 2; ++ms) {
      const _Float16* ap = &ldsA[ms * 16 + rowA][0];
      afrag[ms] = cat8(*(const v8h*)(ap + koff), *(const v8h*)(ap + 16 + koff));
    }
#pragma unroll
    for (int j = 0; j < 3; ++j) {
      const int n  = nb0 + j * 16 + (lane & 15);
      const int kb = kt + ((lane < 16) ? 0 : 8);
      const _Float16* bp = Wcat + (size_t)n * D_MODEL + kb;
      v16h bfrag = cat8(*(const v8h*)bp, *(const v8h*)(bp + 16));
#pragma unroll
      for (int ms = 0; ms < 2; ++ms) {
        acc[ms][j] = __builtin_amdgcn_wmma_f32_16x16x32_f16(
            false, afrag[ms], false, bfrag, (short)0, acc[ms][j], false, false);
      }
    }
    __syncthreads();
  }

#pragma unroll
  for (int ms = 0; ms < 2; ++ms) {
#pragma unroll
    for (int j = 0; j < 3; ++j) {
      const int col  = nb0 + j * 16 + (lane & 15);
      const float bb = bcat[col];
#pragma unroll
      for (int r = 0; r < 8; ++r) {
        const int row = r0 + ms * 16 + ((lane < 16) ? 0 : 8) + r;
        QKV[(size_t)row * N3 + col] = (_Float16)(acc[ms][j][r] + bb);
      }
    }
  }
}

// ---------------------------------------------------------------------------
// Kernel 3: S_b += scale * K_b^T @ V_b   (split-K over sequence, fp32 atomics)
// grid = 4 batches x 8 splits; K/V chunk transposed into LDS as [d][t].
// ---------------------------------------------------------------------------
__global__ void __launch_bounds__(256) kv_outer_kernel(const _Float16* __restrict__ QKV,
                                                       float* __restrict__ Sacc) {
  __shared__ _Float16 kT[DIMS][40];
  __shared__ _Float16 vT[DIMS][40];
  const int tid   = threadIdx.x;
  const int lane  = tid & 31;
  const int wave  = tid >> 5;
  const int b     = blockIdx.x & 3;
  const int split = blockIdx.x >> 2;
  const int tspan = SEQ / SPLITK;             // 512
  const int t0    = split * tspan;
  const int rowA  = wave * 16 + (lane & 15);  // dk row
  const int koff  = (lane < 16) ? 0 : 8;
  const int kb    = (lane < 16) ? 0 : 8;

  v8f acc[8] = {};

  for (int c = 0; c < tspan; c += TCHUNK) {
    const size_t base = ((size_t)b * SEQ + t0 + c) * N3;
    for (int i = tid; i < TCHUNK * DIMS; i += 256) {
      int tl = i >> 7;                        // 0..31
      int d  = i & 127;
      kT[d][tl] = QKV[base + (size_t)tl * N3 + DIMS + d];
      vT[d][tl] = QKV[base + (size_t)tl * N3 + 2 * DIMS + d];
    }
    __syncthreads();

    const _Float16* ap = &kT[rowA][0];
    v16h afrag = cat8(*(const v8h*)(ap + koff), *(const v8h*)(ap + 16 + koff));
#pragma unroll
    for (int j = 0; j < 8; ++j) {
      const _Float16* bp = &vT[j * 16 + (lane & 15)][0] + kb;
      v16h bfrag = cat8(*(const v8h*)bp, *(const v8h*)(bp + 16));
      acc[j] = __builtin_amdgcn_wmma_f32_16x16x32_f16(
          false, afrag, false, bfrag, (short)0, acc[j], false, false);
    }
    __syncthreads();
  }

#pragma unroll
  for (int j = 0; j < 8; ++j) {
#pragma unroll
    for (int r = 0; r < 8; ++r) {
      const int dk = wave * 16 + ((lane < 16) ? 0 : 8) + r;
      const int dv = j * 16 + (lane & 15);
      atomicAdd(&Sacc[((size_t)b * DIMS + dk) * DIMS + dv], acc[j][r] * SCALE);
    }
  }
}

// ---------------------------------------------------------------------------
// Kernel 4: S fp32 [dk][dv] -> f16 transposed [dv][dk] (B-operand layout)
// ---------------------------------------------------------------------------
__global__ void sconv_kernel(const float* __restrict__ Sacc, _Float16* __restrict__ S16T) {
  int gid = blockIdx.x * blockDim.x + threadIdx.x;
  if (gid < BATCH * DIMS * DIMS) {
    int b   = gid >> 14;
    int rem = gid & (DIMS * DIMS - 1);
    int dk  = rem >> 7;
    int dv  = rem & 127;
    S16T[((size_t)b * DIMS + dv) * DIMS + dk] = (_Float16)Sacc[gid];
  }
}

// ---------------------------------------------------------------------------
// Kernel 5: out_b = q_b @ S'_b   ([4096x128]*[128x128] per batch, fp32 out).
// S'_b (32KB f16) is DMA'd into LDS once per block by the Tensor Data Mover,
// with TDM padding: 4 DWORDs every 64 DWORDs (one 128-half row) -> row stride
// 272B, keeping v8h fragment loads 16B-aligned and bank-conflict-free.
// ---------------------------------------------------------------------------
__global__ void __launch_bounds__(256) out_kernel(const _Float16* __restrict__ QKV,
                                                  const _Float16* __restrict__ S16T,
                                                  float* __restrict__ out) {
  __shared__ _Float16 ldsS[DIMS * SROW];      // 128 * 272B = 34816 B
  const int tid  = threadIdx.x;
  const int lane = tid & 31;
  const int wave = tid >> 5;
  const int b    = blockIdx.x >> 5;
  const int mt   = blockIdx.x & 31;
  const int r0   = mt * 128 + wave * 16;
  const int koff = (lane < 16) ? 0 : 8;
  const int kb   = (lane < 16) ? 0 : 8;
  const size_t qrow = ((size_t)b * SEQ + r0 + (lane & 15)) * N3;   // q cols 0..127

  if (wave == 0) {
    // ---- Tensor DMA descriptor (D#), packed per CDNA5 ISA §8.3/§8.4 ----
    const unsigned long long ga =
        (unsigned long long)(uintptr_t)(S16T + (size_t)b * DIMS * DIMS);
    const unsigned int la = (unsigned int)(uintptr_t)(&ldsS[0]);   // LDS byte addr

    v4u g0;                                  // group0 (128b)
    g0[0] = 1u;                              // count=1, is_restore=0, gather off
    g0[1] = la;                              // lds_addr
    g0[2] = (unsigned int)ga;                // global_addr[31:0]
    g0[3] = (unsigned int)((ga >> 32) & 0x1FFFFFFu) | 0x80000000u; // addr[56:32]|type=2

    v8i g1;                                  // group1 (256b)
    g1[0] = (int)((1u << 16)    // data_size = 2 bytes
                | (1u << 20)    // pad_enable
                | (5u << 22)    // pad_interval: 64 DWORDs (one 128-half row)
                | (3u << 25));  // pad_amount: 4 DWORDs
    g1[1] = (int)(128u << 16);  // tensor_dim0[15:0] = 128   (bits 63:48)
    g1[2] = (int)(128u << 16);  // tensor_dim1[15:0] = 128   (bits 95:80)
    g1[3] = (int)(128u << 16);  // tile_dim0 = 128           (bits 127:112)
    g1[4] = 128;                // tile_dim1 = 128, tile_dim2 = 0
    g1[5] = 128;                // tensor_dim0_stride[31:0] = 128 elements
    g1[6] = 0;                  // stride0[47:32]=0, tensor_dim1_stride[15:0]=0
    g1[7] = 0;

    v4i gz = {};                // groups 2/3 unused (2D tensor)
#if __clang_major__ >= 23
    v8i gz8 = {};
    __builtin_amdgcn_tensor_load_to_lds(g0, g1, gz, gz, gz8, 0);
#else
    __builtin_amdgcn_tensor_load_to_lds(g0, g1, gz, gz, 0);
#endif
    __builtin_amdgcn_s_wait_tensorcnt((short)0);
  }
  __syncthreads();

  v8f acc[8] = {};

#pragma unroll
  for (int kt = 0; kt < DIMS; kt += 32) {
    v16h afrag = cat8(*(const v8h*)(QKV + qrow + kt + koff),
                      *(const v8h*)(QKV + qrow + kt + 16 + koff));
#pragma unroll
    for (int j = 0; j < 8; ++j) {
      const _Float16* bp = &ldsS[(size_t)(j * 16 + (lane & 15)) * SROW + kt + kb];
      v16h bfrag = cat8(*(const v8h*)bp, *(const v8h*)(bp + 16));
      acc[j] = __builtin_amdgcn_wmma_f32_16x16x32_f16(
          false, afrag, false, bfrag, (short)0, acc[j], false, false);
    }
  }

#pragma unroll
  for (int j = 0; j < 8; ++j) {
#pragma unroll
    for (int r = 0; r < 8; ++r) {
      const int s = r0 + ((lane < 16) ? 0 : 8) + r;
      out[(((size_t)b * SEQ) + s) * DIMS + j * 16 + (lane & 15)] = acc[j][r];
    }
  }
}

// ---------------------------------------------------------------------------
extern "C" void kernel_launch(void* const* d_in, const int* in_sizes, int n_in,
                              void* d_out, int out_size, void* d_ws, size_t ws_size,
                              hipStream_t stream) {
  const float* x  = (const float*)d_in[0];
  const float* Wq = (const float*)d_in[1];
  const float* bq = (const float*)d_in[2];
  const float* Wk = (const float*)d_in[3];
  const float* bk = (const float*)d_in[4];
  const float* Wv = (const float*)d_in[5];
  const float* bv = (const float*)d_in[6];
  float* out = (float*)d_out;

  char* ws = (char*)d_ws;
  const size_t QKV_BYTES  = (size_t)BATCH * SEQ * N3 * 2;     // 12,582,912
  const size_t WCAT_BYTES = (size_t)N3 * D_MODEL * 2;         //    786,432
  const size_t BCAT_BYTES = 2048;                             // 384*4 padded
  const size_t SACC_BYTES = (size_t)BATCH * DIMS * DIMS * 4;  //    262,144

  _Float16* QKV  = (_Float16*)(ws);
  _Float16* Wcat = (_Float16*)(ws + QKV_BYTES);
  float*    bcat = (float*)   (ws + QKV_BYTES + WCAT_BYTES);
  float*    Sacc = (float*)   (ws + QKV_BYTES + WCAT_BYTES + BCAT_BYTES);
  _Float16* S16T = (_Float16*)(ws + QKV_BYTES + WCAT_BYTES + BCAT_BYTES + SACC_BYTES);

  prep_kernel<<<(N3 * D_MODEL + 255) / 256, 256, 0, stream>>>(Wq, Wk, Wv, bq, bk, bv,
                                                              Wcat, bcat, Sacc);
  qkv_kernel<<<(BATCH * SEQ) / 32, 256, 0, stream>>>(x, Wcat, bcat, QKV);
  kv_outer_kernel<<<BATCH * SPLITK, 256, 0, stream>>>(QKV, Sacc);
  sconv_kernel<<<(BATCH * DIMS * DIMS) / 256, 256, 0, stream>>>(Sacc, S16T);
  out_kernel<<<BATCH * (SEQ / 128), 256, 0, stream>>>(QKV, S16T, out);
}
